// Transformers_65369402245731
// MI455X (gfx1250) — matrix operand order repaired and checked by reading, not compiled, not run on hardware
//
#include <hip/hip_runtime.h>

// ---------------------------------------------------------------------------
// Transformer encoder-decoder forward for MI455X (gfx1250, wave32, WMMA).
// All GEMMs run on v_wmma_f32_16x16x32_bf16 (bf16 in, f32 accumulate).
// GEMM: 128x64 block tile, BK=32, double-buffered LDS, 8 WMMA / k-step / wave,
// tile staging via GLOBAL_LOAD_ASYNC_TO_LDS (ASYNCcnt) where possible.
// ---------------------------------------------------------------------------

#define D_MODEL 1024
#define N_HEAD  16
#define D_HEAD  64
#define FFN     4096
#define BATCH   2
#define SEQ     512          // S == T == 512
#define VOCAB   16000
#define NLAYER  4
#define NTOK    (BATCH * SEQ)   // 1024 rows for all "big" GEMMs

typedef __attribute__((ext_vector_type(16))) __bf16   v16bf;
typedef __attribute__((ext_vector_type(8)))  float    v8f;
typedef __attribute__((ext_vector_type(4)))  unsigned u32x4;
typedef __attribute__((ext_vector_type(2)))  unsigned u32x2;

union FragBF { v16bf v; u32x4 q[2]; };

#if defined(__HIP_DEVICE_COMPILE__) && __has_builtin(__builtin_amdgcn_global_load_async_to_lds_b64)
#define USE_ASYNC_LDS 1
typedef __attribute__((ext_vector_type(2))) int v2i_t;   // b64 payload type
typedef __attribute__((address_space(1))) v2i_t* glb_v2i;
typedef __attribute__((address_space(3))) v2i_t* lds_v2i;
#else
#define USE_ASYNC_LDS 0
#endif

__device__ __forceinline__ void wait_async0() {
#if defined(__HIP_DEVICE_COMPILE__)
#if __has_builtin(__builtin_amdgcn_s_wait_asynccnt)
  __builtin_amdgcn_s_wait_asynccnt(0);
#else
  asm volatile("s_wait_asynccnt 0" ::: "memory");
#endif
#endif
}

__device__ __forceinline__ unsigned short f32_to_bf16(float f) {
  unsigned u = __float_as_uint(f);
  u += 0x7FFFu + ((u >> 16) & 1u);          // round-to-nearest-even
  return (unsigned short)(u >> 16);
}

// ---------------------------------------------------------------------------
// Stage one 128x32 A tile (row-major) and one 32x64 B tile (transposed into
// sb[n][k]) into LDS. Contiguous copies go through the CDNA5 async-DMA path
// (GLOBAL_LOAD_ASYNC_TO_LDS_B64, completion via s_wait_asynccnt before the
// barrier); the row-major-B transpose is a scatter, staged load-then-store so
// global loads pipeline ahead of the dependent LDS writes.
// ---------------------------------------------------------------------------
__device__ __forceinline__ void stage_tiles(
    unsigned short (*sa)[40], unsigned short (*sb)[40],
    const unsigned short* __restrict__ A, const unsigned short* __restrict__ B,
    long tileM, long tileN, long lda, long ldb, int k0, int transB, int tid)
{
  // ---- A tile: 128 rows x 32 halfs, 8B per lane copy (always contiguous)
#if USE_ASYNC_LDS
#pragma unroll
  for (int t = 0; t < 8; ++t) {
    int i = tid + t * 128;
    int r = i >> 3;
    int c = (i & 7) << 2;
    __builtin_amdgcn_global_load_async_to_lds_b64(
        (glb_v2i)(A + (tileM + r) * lda + (k0 + c)), (lds_v2i)&sa[r][c], 0, 0);
  }
#else
  {
    u32x2 av[8];
#pragma unroll
    for (int t = 0; t < 8; ++t) {
      int i = tid + t * 128;
      av[t] = *(const u32x2*)(A + (tileM + (i >> 3)) * lda + (k0 + ((i & 7) << 2)));
    }
#pragma unroll
    for (int t = 0; t < 8; ++t) {
      int i = tid + t * 128;
      *(u32x2*)&sa[i >> 3][(i & 7) << 2] = av[t];
    }
  }
#endif

  if (transB) {                            // B(k,n) at B + n*ldb + k : contiguous
#if USE_ASYNC_LDS
#pragma unroll
    for (int t = 0; t < 4; ++t) {
      int i = tid + t * 128;
      int n = i >> 3;
      int c = (i & 7) << 2;
      __builtin_amdgcn_global_load_async_to_lds_b64(
          (glb_v2i)(B + (tileN + n) * ldb + (k0 + c)), (lds_v2i)&sb[n][c], 0, 0);
    }
#else
    u32x2 bv[4];
#pragma unroll
    for (int t = 0; t < 4; ++t) {
      int i = tid + t * 128;
      bv[t] = *(const u32x2*)(B + (tileN + (i >> 3)) * ldb + (k0 + ((i & 7) << 2)));
    }
#pragma unroll
    for (int t = 0; t < 4; ++t) {
      int i = tid + t * 128;
      *(u32x2*)&sb[i >> 3][(i & 7) << 2] = bv[t];
    }
#endif
  } else {                                 // row-major B: transpose-scatter into LDS
    u32x2 bv[4];
#pragma unroll
    for (int t = 0; t < 4; ++t) {
      int i = tid + t * 128;
      bv[t] = *(const u32x2*)(B + (long)(k0 + (i >> 4)) * ldb + tileN + ((i & 15) << 2));
    }
#pragma unroll
    for (int t = 0; t < 4; ++t) {
      int i  = tid + t * 128;
      int kk = i >> 4;
      int n4 = (i & 15) << 2;
      const unsigned short* p = (const unsigned short*)&bv[t];
      sb[n4 + 0][kk] = p[0];
      sb[n4 + 1][kk] = p[1];
      sb[n4 + 2][kk] = p[2];
      sb[n4 + 3][kk] = p[3];
    }
  }
}

// ---------------------------------------------------------------------------
// Generic batched GEMM: C[M,N] = A[M,K](bf16) * B[K,N](bf16) (+bias, relu)
// Block tile 128x64, BK=32, 128 threads (4 waves); wave owns a 32x64 slab
// -> 2 A-frags x 4 B-frags -> 8 v_wmma per k-step. Double-buffered LDS.
// transB: B element (k,n) is at B + n*ldb + k (Q*K^T and tied logits).
// Batch z: offsets = (z/bdiv)*s?1 + (z%bdiv)*s?2 (element strides).
// All M mult of 128, N mult of 64, K mult of 32 in this net -> no edge code.
// ---------------------------------------------------------------------------
__global__ __launch_bounds__(128) void wmma_gemm_bf16_kernel(
    const unsigned short* __restrict__ A, const unsigned short* __restrict__ B,
    void* __restrict__ Cv, const float* __restrict__ bias,
    int K, long lda, long ldb, long ldc,
    int transB, int relu, int storeb16,
    int bdiv, long sA1, long sA2, long sB1, long sB2, long sC1, long sC2)
{
  const int z  = blockIdx.z;
  const int z1 = z / bdiv, z2 = z % bdiv;
  A += z1 * sA1 + z2 * sA2;
  B += z1 * sB1 + z2 * sB2;
  const long coff = z1 * sC1 + z2 * sC2;

  __shared__ __attribute__((aligned(16))) unsigned short sa[2][128][40]; // [m][k]
  __shared__ __attribute__((aligned(16))) unsigned short sb[2][64][40];  // [n][k]

  const int tid  = threadIdx.x;
  const int wave = tid >> 5;
  const int lane = tid & 31;
  const int hl   = lane >> 4;     // lane half (0/1)
  const int l15  = lane & 15;
  const long tileM = (long)blockIdx.y * 128;
  const long tileN = (long)blockIdx.x * 64;

  v8f acc[2][4] = {};

  // prologue: stage k-tile 0 into buffer 0
  stage_tiles(sa[0], sb[0], A, B, tileM, tileN, lda, ldb, 0, transB, tid);
  wait_async0();
  __syncthreads();

  int buf = 0;
  for (int k0 = 0; k0 < K; k0 += 32) {
    const int nxt = buf ^ 1;
    if (k0 + 32 < K)   // overlap: async-stage next tile while computing current
      stage_tiles(sa[nxt], sb[nxt], A, B, tileM, tileN, lda, ldb, k0 + 32, transB, tid);
    if (k0 + 64 < K) { // pull the tile after that toward L2/WGP$
      __builtin_prefetch((const void*)(A + (tileM + (tid >> 3)) * lda + (k0 + 64 + ((tid & 7) << 2))), 0, 1);
      __builtin_prefetch((const void*)(B + (transB ? ((tileN + (tid >> 3)) * ldb + k0 + 64)
                                                   : ((long)(k0 + 64 + (tid >> 4)) * ldb + tileN))), 0, 1);
    }

    // ---- fragments from current buffer
    // A (16x32): j=0..7 -> K=8*hl+j ; j=8..15 -> K=16+8*hl+(j-8)
    FragBF af[2];
#pragma unroll
    for (int r = 0; r < 2; ++r) {
      const unsigned short* row = &sa[buf][wave * 32 + r * 16 + l15][0];
      af[r].q[0] = *(const u32x4*)(row + hl * 8);
      af[r].q[1] = *(const u32x4*)(row + 16 + hl * 8);
    }
    // B (32x16): lane=n, j=0..7 -> K=8*hl+j ; j=8..15 -> K=16+8*hl+(j-8)
    FragBF bfr[4];
#pragma unroll
    for (int n = 0; n < 4; ++n) {
      const unsigned short* row = &sb[buf][n * 16 + l15][0];
      bfr[n].q[0] = *(const u32x4*)(row + hl * 8);
      bfr[n].q[1] = *(const u32x4*)(row + 16 + hl * 8);
    }
#pragma unroll
    for (int r = 0; r < 2; ++r)
#pragma unroll
      for (int n = 0; n < 4; ++n)
        acc[r][n] = __builtin_amdgcn_wmma_f32_16x16x32_bf16(
            false, af[r].v, false, bfr[n].v, (short)0, acc[r][n], false, false);

    wait_async0();     // async tile DMA into next buffer complete
    __syncthreads();   // LDS stores visible; current-buffer reads done
    buf = nxt;
  }

  // ---- epilogue: C/D layout: VGPR e -> row 8*hl + e, col = l15
#pragma unroll
  for (int r = 0; r < 2; ++r) {
    const long rowBase = tileM + wave * 32 + r * 16 + hl * 8;
#pragma unroll
    for (int nt = 0; nt < 4; ++nt) {
      const long col = tileN + nt * 16 + l15;
      const float bv = bias ? bias[col] : 0.0f;
#pragma unroll
      for (int e = 0; e < 8; ++e) {
        float v = acc[r][nt][e] + bv;
        if (relu) v = fmaxf(v, 0.0f);
        const long idx = coff + (rowBase + e) * ldc + col;
        if (storeb16) ((unsigned short*)Cv)[idx] = f32_to_bf16(v);
        else          ((float*)Cv)[idx] = v;
      }
    }
  }
}

// ---------------------------------------------------------------------------
// Elementwise / reduction kernels
// ---------------------------------------------------------------------------
__global__ __launch_bounds__(256) void cast_f32_bf16_kernel(
    const float* __restrict__ in, unsigned short* __restrict__ out, long n)
{
  long i = (long)blockIdx.x * 256 + threadIdx.x;
  const long stride = (long)gridDim.x * 256;
  for (; i < n; i += stride) out[i] = f32_to_bf16(in[i]);
}

// Wq[H][D][DK] -> out[D][H*DK]  (so all heads become one D x D GEMM operand)
__global__ __launch_bounds__(256) void cast_qkv_kernel(
    const float* __restrict__ w, unsigned short* __restrict__ out)
{
  const int i  = blockIdx.x * 256 + threadIdx.x;   // grid sized exactly H*D*DK
  const int kk = i & (D_HEAD - 1);
  const int r  = i >> 6;           // h*D + d
  const int d  = r & (D_MODEL - 1);
  const int h  = r >> 10;
  out[(long)d * D_MODEL + h * D_HEAD + kk] = f32_to_bf16(w[i]);
}

__global__ __launch_bounds__(256) void embed_kernel(
    const int* __restrict__ tok, const float* __restrict__ emb,
    float* __restrict__ outf, unsigned short* __restrict__ outb)
{
  const int bs = blockIdx.x;
  const int s  = bs & (SEQ - 1);
  const long t = tok[bs];
  for (int d = threadIdx.x; d < D_MODEL; d += 256) {
    const int i = d >> 1;
    const float ang = (float)s * powf(10000.0f, -2.0f * (float)i / (float)D_MODEL);
    const float pe  = (d & 1) ? cosf(ang) : sinf(ang);
    const float v   = emb[t * D_MODEL + d] + pe;
    outf[(long)bs * D_MODEL + d] = v;
    outb[(long)bs * D_MODEL + d] = f32_to_bf16(v);
  }
}

// Row softmax over ncols with optional causal mask; emits bf16 probabilities.
__global__ __launch_bounds__(256) void softmax_kernel(
    const float* __restrict__ sc, unsigned short* __restrict__ pb,
    int ncols, float scale, int causal)
{
  const long row = (long)blockIdx.y * gridDim.x + blockIdx.x;
  const int  q   = blockIdx.x;
  const float* s0 = sc + row * ncols;
  const int limit = causal ? (q + 1) : ncols;
  __shared__ float sh[256];

  float m = -1e30f;
  for (int i = threadIdx.x; i < limit; i += 256) m = fmaxf(m, s0[i] * scale);
  sh[threadIdx.x] = m; __syncthreads();
  for (int st = 128; st > 0; st >>= 1) {
    if (threadIdx.x < st) sh[threadIdx.x] = fmaxf(sh[threadIdx.x], sh[threadIdx.x + st]);
    __syncthreads();
  }
  m = sh[0]; __syncthreads();

  float sum = 0.0f;
  for (int i = threadIdx.x; i < limit; i += 256) sum += expf(s0[i] * scale - m);
  sh[threadIdx.x] = sum; __syncthreads();
  for (int st = 128; st > 0; st >>= 1) {
    if (threadIdx.x < st) sh[threadIdx.x] += sh[threadIdx.x + st];
    __syncthreads();
  }
  const float inv = 1.0f / sh[0];

  for (int i = threadIdx.x; i < ncols; i += 256) {
    const float p = (i < limit) ? expf(s0[i] * scale - m) * inv : 0.0f;
    pb[row * ncols + i] = f32_to_bf16(p);
  }
}

// out = LayerNorm(res + c + bias) with unbiased variance (ddof=1), eps=1e-12.
// Writes fp32 (next residual) and bf16 (next GEMM A operand).
__global__ __launch_bounds__(256) void ln_res_kernel(
    const float* __restrict__ res, const float* __restrict__ c,
    const float* __restrict__ bias, const float* __restrict__ g,
    const float* __restrict__ b, float* __restrict__ outf,
    unsigned short* __restrict__ outb)
{
  const long row = blockIdx.x;
  const float* r0 = res + row * D_MODEL;
  const float* c0 = c   + row * D_MODEL;
  float tv[4];
  float s = 0.0f, ss = 0.0f;
  for (int i = threadIdx.x, j = 0; i < D_MODEL; i += 256, ++j) {
    const float t = r0[i] + c0[i] + (bias ? bias[i] : 0.0f);
    tv[j] = t; s += t; ss += t * t;
  }
  __shared__ float sh1[256], sh2[256];
  sh1[threadIdx.x] = s; sh2[threadIdx.x] = ss; __syncthreads();
  for (int st = 128; st > 0; st >>= 1) {
    if (threadIdx.x < st) {
      sh1[threadIdx.x] += sh1[threadIdx.x + st];
      sh2[threadIdx.x] += sh2[threadIdx.x + st];
    }
    __syncthreads();
  }
  const float mean = sh1[0] / (float)D_MODEL;
  const float var  = (sh2[0] - (float)D_MODEL * mean * mean) / (float)(D_MODEL - 1);
  const float rstd = 1.0f / sqrtf(var + 1e-12f);
  for (int i = threadIdx.x, j = 0; i < D_MODEL; i += 256, ++j) {
    const float v = (tv[j] - mean) * rstd * g[i] + b[i];
    outf[row * D_MODEL + i] = v;
    outb[row * D_MODEL + i] = f32_to_bf16(v);
  }
}

// ---------------------------------------------------------------------------
// Host orchestration
// ---------------------------------------------------------------------------
static void gemm(hipStream_t st, const void* A, const void* B, void* C,
                 const float* bias, int M, int N, int K,
                 long lda, long ldb, long ldc,
                 bool transB, bool relu, bool b16,
                 int batches = 1, int bdiv = 1,
                 long sA1 = 0, long sA2 = 0, long sB1 = 0, long sB2 = 0,
                 long sC1 = 0, long sC2 = 0)
{
  dim3 g(N / 64, M / 128, batches), blk(128);
  wmma_gemm_bf16_kernel<<<g, blk, 0, st>>>(
      (const unsigned short*)A, (const unsigned short*)B, C, bias,
      K, lda, ldb, ldc, transB ? 1 : 0, relu ? 1 : 0, b16 ? 1 : 0,
      bdiv, sA1, sA2, sB1, sB2, sC1, sC2);
}

static void cast_bf16(hipStream_t st, const float* in, void* out, long n) {
  int blocks = (int)((n + 2047) / 2048);
  cast_f32_bf16_kernel<<<blocks, 256, 0, st>>>(in, (unsigned short*)out, n);
}

static inline void* ws_alloc(char* base, size_t& off, size_t bytes) {
  void* p = base + off;
  off += (bytes + 255) & ~(size_t)255;
  return p;
}

extern "C" void kernel_launch(void* const* d_in, const int* in_sizes, int n_in,
                              void* d_out, int out_size, void* d_ws, size_t ws_size,
                              hipStream_t stream) {
  (void)in_sizes; (void)n_in; (void)out_size; (void)ws_size;

  // --- inputs (setup_inputs dict order, nested dicts flattened in order)
  const int*   src  = (const int*)d_in[0];
  const int*   tgt  = (const int*)d_in[1];
  const float* emb  = (const float*)d_in[2];
  const float* eWq  = (const float*)d_in[3];
  const float* eWk  = (const float*)d_in[4];
  const float* eWv  = (const float*)d_in[5];
  const float* eWo  = (const float*)d_in[6];
  const float* ebo  = (const float*)d_in[7];
  const float* eg1  = (const float*)d_in[8];
  const float* eb1  = (const float*)d_in[9];
  const float* eW1  = (const float*)d_in[10];
  const float* ebf1 = (const float*)d_in[11];
  const float* eW2  = (const float*)d_in[12];
  const float* ebf2 = (const float*)d_in[13];
  const float* eg2  = (const float*)d_in[14];
  const float* eb2  = (const float*)d_in[15];
  const float* dsWq = (const float*)d_in[16];
  const float* dsWk = (const float*)d_in[17];
  const float* dsWv = (const float*)d_in[18];
  const float* dsWo = (const float*)d_in[19];
  const float* dsbo = (const float*)d_in[20];
  const float* dcWq = (const float*)d_in[21];
  const float* dcWk = (const float*)d_in[22];
  const float* dcWv = (const float*)d_in[23];
  const float* dcWo = (const float*)d_in[24];
  const float* dcbo = (const float*)d_in[25];
  const float* dg1  = (const float*)d_in[26];
  const float* db1  = (const float*)d_in[27];
  const float* dg2  = (const float*)d_in[28];
  const float* db2  = (const float*)d_in[29];
  const float* dg3  = (const float*)d_in[30];
  const float* db3  = (const float*)d_in[31];
  const float* dW1  = (const float*)d_in[32];
  const float* dbf1 = (const float*)d_in[33];
  const float* dW2  = (const float*)d_in[34];
  const float* dbf2 = (const float*)d_in[35];

  // --- workspace layout
  char* base = (char*)d_ws; size_t off = 0;
  float*          xf      = (float*)         ws_alloc(base, off, (size_t)NTOK * D_MODEL * 4);
  float*          yf      = (float*)         ws_alloc(base, off, (size_t)NTOK * D_MODEL * 4);
  unsigned short* xb      = (unsigned short*)ws_alloc(base, off, (size_t)NTOK * D_MODEL * 2);
  unsigned short* yb      = (unsigned short*)ws_alloc(base, off, (size_t)NTOK * D_MODEL * 2);
  unsigned short* qb      = (unsigned short*)ws_alloc(base, off, (size_t)NTOK * D_MODEL * 2);
  unsigned short* kb      = (unsigned short*)ws_alloc(base, off, (size_t)NTOK * D_MODEL * 2);
  unsigned short* vb      = (unsigned short*)ws_alloc(base, off, (size_t)NTOK * D_MODEL * 2);
  unsigned short* attnb   = (unsigned short*)ws_alloc(base, off, (size_t)NTOK * D_MODEL * 2);
  unsigned short* hb      = (unsigned short*)ws_alloc(base, off, (size_t)NTOK * FFN * 2);
  float*          cf      = (float*)         ws_alloc(base, off, (size_t)NTOK * D_MODEL * 4);
  float*          scoresf = (float*)         ws_alloc(base, off, (size_t)BATCH * N_HEAD * SEQ * SEQ * 4);
  unsigned short* probsb  = (unsigned short*)ws_alloc(base, off, (size_t)BATCH * N_HEAD * SEQ * SEQ * 2);
  unsigned short* wstage  = (unsigned short*)ws_alloc(base, off, (size_t)D_MODEL * FFN * 2);
  unsigned short* embb    = (unsigned short*)ws_alloc(base, off, (size_t)VOCAB * D_MODEL * 2);

  const long HDD = (long)N_HEAD * D_MODEL * D_HEAD;   // per-layer QKV weight slice
  const long DD  = (long)D_MODEL * D_MODEL;
  const long DF  = (long)D_MODEL * FFN;

  // --- embeddings (token gather + sinusoidal positions), and bf16 vocab table
  cast_bf16(stream, emb, embb, (long)VOCAB * D_MODEL);
  embed_kernel<<<NTOK, 256, 0, stream>>>(src, emb, xf, xb);
  embed_kernel<<<NTOK, 256, 0, stream>>>(tgt, emb, yf, yb);

  // --- one multi-head attention block (writes pre-residual proj into cf)
  auto attention = [&](const unsigned short* qsrc, const unsigned short* kvsrc,
                       const float* Wq_l, const float* Wk_l, const float* Wv_l,
                       const float* Wo_l, bool causal) {
    cast_qkv_kernel<<<4096, 256, 0, stream>>>(Wq_l, wstage);
    gemm(stream, qsrc,  wstage, qb, nullptr, NTOK, D_MODEL, D_MODEL,
         D_MODEL, D_MODEL, D_MODEL, false, false, true);
    cast_qkv_kernel<<<4096, 256, 0, stream>>>(Wk_l, wstage);
    gemm(stream, kvsrc, wstage, kb, nullptr, NTOK, D_MODEL, D_MODEL,
         D_MODEL, D_MODEL, D_MODEL, false, false, true);
    cast_qkv_kernel<<<4096, 256, 0, stream>>>(Wv_l, wstage);
    gemm(stream, kvsrc, wstage, vb, nullptr, NTOK, D_MODEL, D_MODEL,
         D_MODEL, D_MODEL, D_MODEL, false, false, true);

    // scores[b,h] = Q_bh (512xDK) * K_bh^T : batched over B*H
    gemm(stream, qb, kb, scoresf, nullptr, SEQ, SEQ, D_HEAD,
         D_MODEL, D_MODEL, SEQ, /*transB=*/true, false, false,
         BATCH * N_HEAD, N_HEAD,
         (long)SEQ * D_MODEL, D_HEAD,           // A: b -> S*D, h -> DK
         (long)SEQ * D_MODEL, D_HEAD,           // B: same layout
         (long)N_HEAD * SEQ * SEQ, (long)SEQ * SEQ);

    softmax_kernel<<<dim3(SEQ, BATCH * N_HEAD), 256, 0, stream>>>(
        scoresf, probsb, SEQ, 0.125f, causal ? 1 : 0);

    // attn[b,h] = P (512x512) * V_bh (512xDK), stored bf16 as [b,s,h,dk]
    gemm(stream, probsb, vb, attnb, nullptr, SEQ, D_HEAD, SEQ,
         SEQ, D_MODEL, D_MODEL, false, false, true,
         BATCH * N_HEAD, N_HEAD,
         (long)N_HEAD * SEQ * SEQ, (long)SEQ * SEQ,
         (long)SEQ * D_MODEL, D_HEAD,
         (long)SEQ * D_MODEL, D_HEAD);

    // output projection (bias folded into following LayerNorm)
    cast_bf16(stream, Wo_l, wstage, DD);
    gemm(stream, attnb, wstage, cf, nullptr, NTOK, D_MODEL, D_MODEL,
         D_MODEL, D_MODEL, D_MODEL, false, false, false);
  };

  // --- encoder stack
  for (int l = 0; l < NLAYER; ++l) {
    attention(xb, xb, eWq + l * HDD, eWk + l * HDD, eWv + l * HDD, eWo + l * DD, false);
    ln_res_kernel<<<NTOK, 256, 0, stream>>>(xf, cf, ebo + l * D_MODEL,
                                            eg1 + l * D_MODEL, eb1 + l * D_MODEL, xf, xb);
    cast_bf16(stream, eW1 + l * DF, wstage, DF);
    gemm(stream, xb, wstage, hb, ebf1 + l * FFN, NTOK, FFN, D_MODEL,
         D_MODEL, FFN, FFN, false, /*relu=*/true, /*b16=*/true);
    cast_bf16(stream, eW2 + l * DF, wstage, DF);
    gemm(stream, hb, wstage, cf, nullptr, NTOK, D_MODEL, FFN,
         FFN, D_MODEL, D_MODEL, false, false, false);
    ln_res_kernel<<<NTOK, 256, 0, stream>>>(xf, cf, ebf2 + l * D_MODEL,
                                            eg2 + l * D_MODEL, eb2 + l * D_MODEL, xf, xb);
  }
  // xb now holds bf16 encoder output (cross-attention K/V source)

  // --- decoder stack
  for (int l = 0; l < NLAYER; ++l) {
    attention(yb, yb, dsWq + l * HDD, dsWk + l * HDD, dsWv + l * HDD, dsWo + l * DD, true);
    ln_res_kernel<<<NTOK, 256, 0, stream>>>(yf, cf, dsbo + l * D_MODEL,
                                            dg1 + l * D_MODEL, db1 + l * D_MODEL, yf, yb);
    attention(yb, xb, dcWq + l * HDD, dcWk + l * HDD, dcWv + l * HDD, dcWo + l * DD, false);
    ln_res_kernel<<<NTOK, 256, 0, stream>>>(yf, cf, dcbo + l * D_MODEL,
                                            dg2 + l * D_MODEL, db2 + l * D_MODEL, yf, yb);
    cast_bf16(stream, dW1 + l * DF, wstage, DF);
    gemm(stream, yb, wstage, hb, dbf1 + l * FFN, NTOK, FFN, D_MODEL,
         D_MODEL, FFN, FFN, false, /*relu=*/true, /*b16=*/true);
    cast_bf16(stream, dW2 + l * DF, wstage, DF);
    gemm(stream, hb, wstage, cf, nullptr, NTOK, D_MODEL, FFN,
         FFN, D_MODEL, D_MODEL, false, false, false);
    ln_res_kernel<<<NTOK, 256, 0, stream>>>(yf, cf, dbf2 + l * D_MODEL,
                                            dg3 + l * D_MODEL, db3 + l * D_MODEL, yf, yb);
  }

  // --- tied logits: y (1024 x 1024) @ emb^T (1024 x 16000) -> d_out (fp32)
  gemm(stream, yb, embb, d_out, nullptr, NTOK, VOCAB, D_MODEL,
       D_MODEL, D_MODEL, VOCAB, /*transB=*/true, false, false);
}
